// TKFFTConv_54159537602648
// MI455X (gfx1250) — matrix operand order repaired and checked by reading, not compiled, not run on hardware
//
#include <hip/hip_runtime.h>

typedef _Float16 h16 __attribute__((ext_vector_type(16)));
typedef _Float16 hv8 __attribute__((ext_vector_type(8)));
typedef float    f8  __attribute__((ext_vector_type(8)));
typedef float    f4  __attribute__((ext_vector_type(4)));
typedef float    f2  __attribute__((ext_vector_type(2)));

#define WAVES 4
#define TPB   (WAVES * 32)
#define PI2   6.28318530717958647692f

__device__ __forceinline__ f8 f8zero() {
  f8 z;
#pragma unroll
  for (int i = 0; i < 8; ++i) z[i] = 0.0f;
  return z;
}

__device__ __forceinline__ f8 wmma_h(h16 a, h16 b, f8 c) {
  // D = A(16x32 f16) * B(32x16 f16) + C(16x16 f32)
  return __builtin_amdgcn_wmma_f32_16x16x32_f16(false, a, false, b, (short)0, c,
                                                false, false);
}

// wave-relative LDS byte offset of a __shared__ pointer (for async LDS ops)
__device__ __forceinline__ unsigned lds_off(const void* p) {
  return (unsigned)(unsigned long long)(__attribute__((address_space(3))) const void*)p;
}

// ---- operand loaders (contiguous k = i + 16*(lane/16), wave32) ----
// A 16x32 f16: M = lane%16, K(i) = i + 16*(lane/16)  -> two contiguous b128
__device__ __forceinline__ h16 loadA_h(const _Float16* P, int bi, int lm, int half) {
  int r = bi * 16 + lm;
  hv8 lo = *(const hv8*)(P + r * 32 + 16 * half);
  hv8 hi = *(const hv8*)(P + r * 32 + 16 * half + 8);
  return __builtin_shufflevector(lo, hi, 0, 1, 2, 3, 4, 5, 6, 7, 8, 9, 10, 11, 12, 13, 14, 15);
}

// B 32x16 f16: N = lane%16, K(i) = i + 16*(lane/16)
__device__ __forceinline__ h16 loadB_h(const _Float16* P, int bj, int lm, int half) {
  int c = bj * 16 + lm;
  h16 o;
#pragma unroll
  for (int i = 0; i < 16; ++i) o[i] = P[(i + 16 * half) * 32 + c];
  return o;
}

// B operand from a real f32 row-major plane
__device__ __forceinline__ h16 loadB_r(const float* P, int bj, int lm, int half) {
  int c = bj * 16 + lm;
  h16 o;
#pragma unroll
  for (int i = 0; i < 16; ++i) o[i] = (_Float16)P[(i + 16 * half) * 32 + c];
  return o;
}

// A operand pair from interleaved-complex f32 plane
__device__ __forceinline__ void loadA_c(const float* X, int bi, int lm, int half,
                                        h16& ar, h16& ai) {
  int r = bi * 16 + lm;
#pragma unroll
  for (int i = 0; i < 16; ++i) {
    int kk = i + 16 * half;
    f2 v = *(const f2*)(X + 2 * (r * 32 + kk));
    ar[i] = (_Float16)v.x;
    ai[i] = (_Float16)v.y;
  }
}

// B operand pair from interleaved-complex f32 plane
__device__ __forceinline__ void loadB_c(const float* X, int bj, int lm, int half,
                                        h16& br, h16& bim) {
  int c = bj * 16 + lm;
#pragma unroll
  for (int i = 0; i < 16; ++i) {
    f2 v = *(const f2*)(X + 2 * ((i + 16 * half) * 32 + c));
    br[i] = (_Float16)v.x;
    bim[i] = (_Float16)v.y;
  }
}

// acc (C layout) * T (interleaved complex) -> X plane (interleaved complex f32)
__device__ __forceinline__ void cmul_store(const f8 aR[2][2], const f8 aI[2][2],
                                           const float* __restrict__ T, float* X,
                                           int lm, int half) {
#pragma unroll
  for (int bi = 0; bi < 2; ++bi)
#pragma unroll
    for (int bj = 0; bj < 2; ++bj)
#pragma unroll
      for (int v = 0; v < 8; ++v) {
        int idx = (bi * 16 + v + 8 * half) * 32 + bj * 16 + lm;
        f2 t = *(const f2*)(T + 2 * idx);
        float xr = aR[bi][bj][v], xi = aI[bi][bj][v];
        f2 o;
        o.x = xr * t.x - xi * t.y;
        o.y = xr * t.y + xi * t.x;
        *(f2*)(X + 2 * idx) = o;
      }
}

// ------------------ kernel 1: filter spectrum kfT -> d_ws ------------------
__global__ __launch_bounds__(TPB) void fftconv_kf(const float* __restrict__ kin,
                                                  float* __restrict__ kf, int H) {
  __shared__ __align__(16) _Float16 sFr[1024], sFi[1024], sFiN[1024];
  __shared__ __align__(16) float sTwN[2048];  // exp(-2pi i rc/1024), no 1/N
  __shared__ __align__(16) float sX[WAVES][2048];

  const int tid = threadIdx.x;
  for (int idx = tid; idx < 1024; idx += TPB) {
    int r = idx >> 5, c = idx & 31;
    float s32, c32, sN, cN;
    __sincosf(-PI2 * (float)((r * c) & 31) * (1.0f / 32.0f), &s32, &c32);
    __sincosf(-PI2 * (float)(r * c) * (1.0f / 1024.0f), &sN, &cN);
    sFr[idx] = (_Float16)c32;
    sFi[idx] = (_Float16)s32;
    sFiN[idx] = (_Float16)(-s32);
    sTwN[2 * idx] = cN;
    sTwN[2 * idx + 1] = sN;
  }
  __syncthreads();

  const int w = tid >> 5, lane = tid & 31, lm = lane & 15, half = lane >> 4;
  const int h = blockIdx.x * WAVES + w;
  if (h >= H) return;
  float* myX = sX[w];
  const float* __restrict__ kp = kin + (size_t)h * 1024;

#pragma unroll
  for (int i = 0; i < 8; ++i)
    *(f4*)(myX + (i * 32 + lane) * 4) = *(const f4*)(kp + (i * 32 + lane) * 4);

  f8 aR[2][2], aI[2][2];

  // X = F @ k   (A complex const, B real)
  {
    h16 bU[2];
#pragma unroll
    for (int bj = 0; bj < 2; ++bj) bU[bj] = loadB_r(myX, bj, lm, half);
#pragma unroll
    for (int bi = 0; bi < 2; ++bi) {
      h16 aFr = loadA_h(sFr, bi, lm, half);
      h16 aFi = loadA_h(sFi, bi, lm, half);
#pragma unroll
      for (int bj = 0; bj < 2; ++bj) {
        aR[bi][bj] = wmma_h(aFr, bU[bj], f8zero());
        aI[bi][bj] = wmma_h(aFi, bU[bj], f8zero());
      }
    }
  }
  cmul_store(aR, aI, sTwN, myX, lm, half);  // twiddle (no 1/N)

  // X = X @ F  (complex x complex)
  h16 bFr[2], bFi[2], bFiN[2];
#pragma unroll
  for (int bj = 0; bj < 2; ++bj) {
    bFr[bj] = loadB_h(sFr, bj, lm, half);
    bFi[bj] = loadB_h(sFi, bj, lm, half);
    bFiN[bj] = loadB_h(sFiN, bj, lm, half);
  }
#pragma unroll
  for (int bi = 0; bi < 2; ++bi) {
    h16 axr, axi;
    loadA_c(myX, bi, lm, half, axr, axi);
#pragma unroll
    for (int bj = 0; bj < 2; ++bj) {
      f8 rr = f8zero(), mm = f8zero();
      rr = wmma_h(axr, bFr[bj], rr);
      rr = wmma_h(axi, bFiN[bj], rr);  // Xr*Fr - Xi*Fi
      mm = wmma_h(axr, bFi[bj], mm);
      mm = wmma_h(axi, bFr[bj], mm);   // Xr*Fi + Xi*Fr
      aR[bi][bj] = rr;
      aI[bi][bj] = mm;
    }
  }

  float* __restrict__ kfp = kf + (size_t)h * 2048;  // interleaved complex [k1][k2]
#pragma unroll
  for (int bi = 0; bi < 2; ++bi)
#pragma unroll
    for (int bj = 0; bj < 2; ++bj)
#pragma unroll
      for (int v = 0; v < 8; ++v) {
        int idx = (bi * 16 + v + 8 * half) * 32 + bj * 16 + lm;
        f2 o;
        o.x = aR[bi][bj][v];
        o.y = aI[bi][bj][v];
        *(f2*)(kfp + 2 * idx) = o;
      }
}

// ------------------ kernel 2: main four-step conv ------------------
__global__ __launch_bounds__(TPB) void fftconv_main(const float* __restrict__ u,
                                                    const float* __restrict__ kf,
                                                    float* __restrict__ out, int H,
                                                    int BH) {
  __shared__ __align__(16) _Float16 sFr[1024], sFi[1024], sFiN[1024];
  __shared__ __align__(16) float sTw[2048];   // exp(-2pi i rc/N)/N
  __shared__ __align__(16) float sTwI[2048];  // exp(+2pi i rc/N)
  __shared__ __align__(16) float sX[WAVES][2048];

  const int tid = threadIdx.x;
  for (int idx = tid; idx < 1024; idx += TPB) {
    int r = idx >> 5, c = idx & 31;
    float s32, c32, sN, cN;
    __sincosf(-PI2 * (float)((r * c) & 31) * (1.0f / 32.0f), &s32, &c32);
    __sincosf(-PI2 * (float)(r * c) * (1.0f / 1024.0f), &sN, &cN);
    sFr[idx] = (_Float16)c32;
    sFi[idx] = (_Float16)s32;
    sFiN[idx] = (_Float16)(-s32);
    sTw[2 * idx] = cN * (1.0f / 1024.0f);
    sTw[2 * idx + 1] = sN * (1.0f / 1024.0f);
    sTwI[2 * idx] = cN;
    sTwI[2 * idx + 1] = -sN;
  }
  __syncthreads();

  const int w = tid >> 5, lane = tid & 31, lm = lane & 15, half = lane >> 4;
  const int tile = blockIdx.x * WAVES + w;  // tile = b*H + h
  if (tile >= BH) return;
  const int h = tile % H;
  float* myX = sX[w];
  const float* __restrict__ up = u + (size_t)tile * 1024;
  const float* __restrict__ kfp = kf + (size_t)h * 2048;
  __builtin_prefetch(kfp, 0, 1);  // global_prefetch of filter spectrum (L2-resident)

  // ---- async stage of input tile (global -> LDS, ASYNCcnt path) ----
  const unsigned myLds = lds_off(myX);
#pragma unroll
  for (int i = 0; i < 8; ++i) {
    unsigned l = myLds + (unsigned)((i * 32 + lane) * 16);
    const float* g = up + (i * 32 + lane) * 4;
    asm volatile("global_load_async_to_lds_b128 %0, %1, off"
                 :: "v"(l), "v"(g) : "memory");
  }

  // hoist all constant-F WMMA operands into registers (overlaps async load)
  h16 aFr[2], aFi[2], bFr[2], bFi[2], bFiN[2];
#pragma unroll
  for (int bi = 0; bi < 2; ++bi) {
    aFr[bi] = loadA_h(sFr, bi, lm, half);
    aFi[bi] = loadA_h(sFi, bi, lm, half);
  }
#pragma unroll
  for (int bj = 0; bj < 2; ++bj) {
    bFr[bj] = loadB_h(sFr, bj, lm, half);
    bFi[bj] = loadB_h(sFi, bj, lm, half);
    bFiN[bj] = loadB_h(sFiN, bj, lm, half);
  }

  asm volatile("s_wait_asynccnt 0" ::: "memory");  // input tile resident in LDS

  f8 aR[2][2], aI[2][2];

  // ---- step 1: X = F @ U (real input -> 2 real matmuls) ----
  {
    h16 bU[2];
#pragma unroll
    for (int bj = 0; bj < 2; ++bj) bU[bj] = loadB_r(myX, bj, lm, half);
#pragma unroll
    for (int bi = 0; bi < 2; ++bi)
#pragma unroll
      for (int bj = 0; bj < 2; ++bj) {
        aR[bi][bj] = wmma_h(aFr[bi], bU[bj], f8zero());
        aI[bi][bj] = wmma_h(aFi[bi], bU[bj], f8zero());
      }
  }
  cmul_store(aR, aI, sTw, myX, lm, half);  // *= tw (with 1/N)

  // ---- step 2: X = X @ F, then *= kfT[h] ----
#pragma unroll
  for (int bi = 0; bi < 2; ++bi) {
    h16 axr, axi;
    loadA_c(myX, bi, lm, half, axr, axi);
#pragma unroll
    for (int bj = 0; bj < 2; ++bj) {
      f8 rr = f8zero(), mm = f8zero();
      rr = wmma_h(axr, bFr[bj], rr);
      rr = wmma_h(axi, bFiN[bj], rr);
      mm = wmma_h(axr, bFi[bj], mm);
      mm = wmma_h(axi, bFr[bj], mm);
      aR[bi][bj] = rr;
      aI[bi][bj] = mm;
    }
  }
  cmul_store(aR, aI, kfp, myX, lm, half);  // *= kfT (global, L2)

  // ---- step 3: X = X @ Finv (Finv = conj F), then *= twinv ----
#pragma unroll
  for (int bi = 0; bi < 2; ++bi) {
    h16 axr, axi;
    loadA_c(myX, bi, lm, half, axr, axi);
#pragma unroll
    for (int bj = 0; bj < 2; ++bj) {
      f8 rr = f8zero(), mm = f8zero();
      rr = wmma_h(axr, bFr[bj], rr);
      rr = wmma_h(axi, bFi[bj], rr);   // Xr*Fr + Xi*Fi
      mm = wmma_h(axr, bFiN[bj], mm);  // Xr*(-Fi) + Xi*Fr
      mm = wmma_h(axi, bFr[bj], mm);
      aR[bi][bj] = rr;
      aI[bi][bj] = mm;
    }
  }
  cmul_store(aR, aI, sTwI, myX, lm, half);  // *= twinv

  // ---- step 4: Y = Re(Finv @ X) -> 2 real matmuls ----
  {
    h16 bXr[2], bXi[2];
#pragma unroll
    for (int bj = 0; bj < 2; ++bj) loadB_c(myX, bj, lm, half, bXr[bj], bXi[bj]);
#pragma unroll
    for (int bi = 0; bi < 2; ++bi)
#pragma unroll
      for (int bj = 0; bj < 2; ++bj) {
        f8 y = f8zero();
        y = wmma_h(aFr[bi], bXr[bj], y);
        y = wmma_h(aFi[bi], bXi[bj], y);  // Fr*Xr + Fi*Xi
        aR[bi][bj] = y;
      }
  }
  // scatter Y row-major (f32) into the staging plane
#pragma unroll
  for (int bi = 0; bi < 2; ++bi)
#pragma unroll
    for (int bj = 0; bj < 2; ++bj)
#pragma unroll
      for (int v = 0; v < 8; ++v)
        myX[(bi * 16 + v + 8 * half) * 32 + bj * 16 + lm] = aR[bi][bj][v];

  // ---- async store of output tile (LDS -> global); S_ENDPGM waits for it ----
  asm volatile("s_wait_dscnt 0" ::: "memory");  // scatter visible to async engine
  float* __restrict__ op = out + (size_t)tile * 1024;
#pragma unroll
  for (int i = 0; i < 8; ++i) {
    unsigned l = myLds + (unsigned)((i * 32 + lane) * 16);
    float* g = op + (i * 32 + lane) * 4;
    asm volatile("global_store_async_from_lds_b128 %0, %1, off"
                 :: "v"(g), "v"(l) : "memory");
  }
}

extern "C" void kernel_launch(void* const* d_in, const int* in_sizes, int n_in,
                              void* d_out, int out_size, void* d_ws, size_t ws_size,
                              hipStream_t stream) {
  const float* u = (const float*)d_in[0];
  const float* k = (const float*)d_in[1];
  float* out = (float*)d_out;
  const int H = in_sizes[1] / 1024;   // 128
  const int BH = in_sizes[0] / 1024;  // B*H = 16384
  float* kf = (float*)d_ws;           // H * 32*32 interleaved complex = 1 MB

  fftconv_kf<<<(H + WAVES - 1) / WAVES, TPB, 0, stream>>>(k, kf, H);
  fftconv_main<<<(BH + WAVES - 1) / WAVES, TPB, 0, stream>>>(u, kf, out, H, BH);
}
